// DGCNN_IA_88802743812386
// MI455X (gfx1250) — matrix operand order repaired and checked by reading, not compiled, not run on hardware
//
#include <hip/hip_runtime.h>
#include <float.h>
#include <math.h>

typedef __attribute__((ext_vector_type(16))) _Float16 v16h;
typedef __attribute__((ext_vector_type(8)))  _Float16 v8h;
typedef __attribute__((ext_vector_type(2)))  _Float16 v2h;
typedef __attribute__((ext_vector_type(8)))  float    v8f;

#define B_SZ  8
#define N_PTS 1024
#define K_NN  20
#define NCOL  (16*K_NN)   // 320 columns = 16 points x 20 neighbors (k-major tiles)
#define CPAD  40          // 32 f16 channels + pad -> 80B row stride (16B aligned)
#define FPAD  36          // 32 f32 channels + pad -> 144B row stride (16B aligned)

static __device__ __forceinline__ v8f wmma_f16(v16h a, v16h b, v8f c) {
  // 8 args: (neg_a, A, neg_b, B, c_mod, C, reuse_a, reuse_b)
  return __builtin_amdgcn_wmma_f32_16x16x32_f16(false, a, false, b, (short)0, c,
                                                false, false);
}

// Build B fragment (32x16 f16): lane col = lane&15, K = ((p<8)?p:p+8) + (lane>>4)*8.
// Tile stored in LDS as [point][channel] rows of CPAD halves.
static __device__ __forceinline__ v16h frag_b_from_lds(const _Float16 (*T)[CPAD],
                                                       int m, int hi) {
  v8h lo = *(const v8h*)&T[m][hi * 8];
  v8h hh = *(const v8h*)&T[m][hi * 8 + 16];
  v16h b;
#pragma unroll
  for (int p = 0; p < 8; ++p) { b[p] = lo[p]; b[p + 8] = hh[p]; }
  return b;
}

// Cooperatively stage a 32(channel) x 16(point) f32 tile into LDS as f16,
// transposed to [point][channel]. Lane = channel row; 4x global_load_b128.
static __device__ __forceinline__ void stage_tile(const float* __restrict__ src,
                                                  int lane, _Float16 (*T)[CPAD]) {
  const float4* p = (const float4*)(src + (size_t)lane * N_PTS);
  float4 r0 = p[0], r1 = p[1], r2 = p[2], r3 = p[3];
  float v[16] = {r0.x, r0.y, r0.z, r0.w, r1.x, r1.y, r1.z, r1.w,
                 r2.x, r2.y, r2.z, r2.w, r3.x, r3.y, r3.z, r3.w};
#pragma unroll
  for (int t = 0; t < 16; ++t) T[t][lane] = (_Float16)v[t];
}

// Same, but keep f32 (for center-subtraction operand).
static __device__ __forceinline__ void stage_tile_f32(const float* __restrict__ src,
                                                      int lane, float (*T)[FPAD]) {
  const float4* p = (const float4*)(src + (size_t)lane * N_PTS);
  float4 r0 = p[0], r1 = p[1], r2 = p[2], r3 = p[3];
  float v[16] = {r0.x, r0.y, r0.z, r0.w, r1.x, r1.y, r1.z, r1.w,
                 r2.x, r2.y, r2.z, r2.w, r3.x, r3.y, r3.z, r3.w};
#pragma unroll
  for (int t = 0; t < 16; ++t) T[t][lane] = v[t];
}

// A operand (16x32 f16 W tile), CIN multiple of 32: two contiguous 8-float runs.
template <int CIN>
static __device__ __forceinline__ v16h load_w_tile(const float* __restrict__ W,
                                                   int row, int c0, int hi) {
  const float* wrow = W + (size_t)row * CIN + c0 + hi * 8;
  float4 w0 = *(const float4*)(wrow);
  float4 w1 = *(const float4*)(wrow + 4);
  float4 w2 = *(const float4*)(wrow + 16);
  float4 w3 = *(const float4*)(wrow + 20);
  float wv[16] = {w0.x, w0.y, w0.z, w0.w, w1.x, w1.y, w1.z, w1.w,
                  w2.x, w2.y, w2.z, w2.w, w3.x, w3.y, w3.z, w3.w};
  v16h a;
#pragma unroll
  for (int p = 0; p < 16; ++p) a[p] = (_Float16)wv[p];
  return a;
}

// ---------------------------------------------------------------------------
// Stage-1 kNN on raw 3-D points: whole 3x1024 slice lives in LDS.
// ---------------------------------------------------------------------------
__global__ __launch_bounds__(256)
void knn3_kernel(const float* __restrict__ x, int* __restrict__ idx) {
  __shared__ float xs0[N_PTS], xs1[N_PTS], xs2[N_PTS], sqs[N_PTS];
  const int b = blockIdx.y;
  const int i = blockIdx.x * 256 + threadIdx.x;
  const float* xb = x + (size_t)b * 3 * N_PTS;
  for (int t = threadIdx.x; t < N_PTS; t += 256) {
    float a0 = xb[t], a1 = xb[N_PTS + t], a2 = xb[2 * N_PTS + t];
    xs0[t] = a0; xs1[t] = a1; xs2[t] = a2;
    sqs[t] = a0 * a0 + a1 * a1 + a2 * a2;
  }
  __syncthreads();
  float xi0 = xs0[i], xi1 = xs1[i], xi2 = xs2[i], sqi = sqs[i];
  float vals[K_NN]; int ids[K_NN];
#pragma unroll
  for (int t = 0; t < K_NN; ++t) { vals[t] = -FLT_MAX; ids[t] = 0; }
  float vmin = -FLT_MAX; int amin = 0;
  for (int j = 0; j < N_PTS; ++j) {
    float d = 2.f * (xi0 * xs0[j] + xi1 * xs1[j] + xi2 * xs2[j]) - sqi - sqs[j];
    if (d > vmin) {
#pragma unroll
      for (int t = 0; t < K_NN; ++t) if (t == amin) { vals[t] = d; ids[t] = j; }
      vmin = vals[0]; amin = 0;
#pragma unroll
      for (int t = 1; t < K_NN; ++t) if (vals[t] < vmin) { vmin = vals[t]; amin = t; }
    }
  }
  int* outp = idx + ((size_t)b * N_PTS + i) * K_NN;
#pragma unroll
  for (int t = 0; t < K_NN; ++t) outp[t] = ids[t];
}

// ---------------------------------------------------------------------------
// Squared norms / column L2-normalize
// ---------------------------------------------------------------------------
__global__ void sqnorm_kernel(const float* __restrict__ x, float* __restrict__ sq, int C) {
  int tid = blockIdx.x * blockDim.x + threadIdx.x;
  if (tid >= B_SZ * N_PTS) return;
  int b = tid / N_PTS, i = tid % N_PTS;
  const float* p = x + (size_t)b * C * N_PTS + i;
  float s = 0.f;
  for (int c = 0; c < C; ++c) { float v = p[(size_t)c * N_PTS]; s += v * v; }
  sq[tid] = s;
}

__global__ void normalize_kernel(const float* __restrict__ x, float* __restrict__ xn, int C) {
  int tid = blockIdx.x * blockDim.x + threadIdx.x;
  if (tid >= B_SZ * N_PTS) return;
  int b = tid / N_PTS, i = tid % N_PTS;
  const float* p = x + (size_t)b * C * N_PTS + i;
  float* q = xn + (size_t)b * C * N_PTS + i;
  float s = 0.f;
  for (int c = 0; c < C; ++c) { float v = p[(size_t)c * N_PTS]; s += v * v; }
  float rn = 1.f / fmaxf(sqrtf(s), 1e-12f);
  for (int c = 0; c < C; ++c) q[(size_t)c * N_PTS] = p[(size_t)c * N_PTS] * rn;
}

// ---------------------------------------------------------------------------
// Gram: G[b,i,j] = sum_c A[b,c,i]*B[b,c,j], C compile-time multiple of 32.
// ---------------------------------------------------------------------------
template <int C>
__global__ __launch_bounds__(32)
void gram_wmma_kernel(const float* __restrict__ A, const float* __restrict__ Bm,
                      float* __restrict__ G) {
  __shared__ __align__(16) _Float16 As[16][CPAD];
  __shared__ __align__(16) _Float16 Bs[16][CPAD];
  const int lane = threadIdx.x;
  const int m = lane & 15, hi = lane >> 4;
  const int i0 = blockIdx.x * 16, j0 = blockIdx.y * 16, b = blockIdx.z;
  const float* Ab = A  + (size_t)b * C * N_PTS;
  const float* Bb = Bm + (size_t)b * C * N_PTS;
  v8f acc;
#pragma unroll
  for (int r = 0; r < 8; ++r) acc[r] = 0.f;
#pragma unroll
  for (int c0 = 0; c0 < C; c0 += 32) {
    stage_tile(Ab + (size_t)c0 * N_PTS + i0, lane, As);
    stage_tile(Bb + (size_t)c0 * N_PTS + j0, lane, Bs);
    __syncthreads();
    // A fragment: row = lane&15, K = ((p<8)?p:p+8)+(lane>>4)*8, from [point][ch] tile
    v8h lo = *(const v8h*)&As[m][hi * 8];
    v8h hh = *(const v8h*)&As[m][hi * 8 + 16];
    v16h a;
#pragma unroll
    for (int p = 0; p < 8; ++p) { a[p] = lo[p]; a[p + 8] = hh[p]; }
    v16h bb = frag_b_from_lds(Bs, m, hi);
    acc = wmma_f16(a, bb, acc);
    __syncthreads();
  }
#pragma unroll
  for (int r = 0; r < 8; ++r) {
    int i = i0 + r + 8 * hi;                 // C/D: M = r + 8*(lane>=16), N = lane&15
    G[((size_t)b * N_PTS + i) * N_PTS + j0 + m] = acc[r];
  }
}

// ---------------------------------------------------------------------------
// Per-row top-K (largest). useDist: d = 2*G - sq_i - sq_j, else raw similarity.
// ---------------------------------------------------------------------------
__global__ void topk_kernel(const float* __restrict__ G, const float* __restrict__ sq,
                            int* __restrict__ idx, int useDist) {
  int tid = blockIdx.x * blockDim.x + threadIdx.x;
  if (tid >= B_SZ * N_PTS) return;
  int b = tid / N_PTS, i = tid % N_PTS;
  const float* row = G + ((size_t)b * N_PTS + i) * N_PTS;
  const float* sqb = sq + (size_t)b * N_PTS;
  float sqi = useDist ? sqb[i] : 0.f;
  float vals[K_NN]; int ids[K_NN];
#pragma unroll
  for (int t = 0; t < K_NN; ++t) { vals[t] = -FLT_MAX; ids[t] = 0; }
  float vmin = -FLT_MAX; int amin = 0;
  for (int j = 0; j < N_PTS; ++j) {
    if ((j & 63) == 0) __builtin_prefetch(row + j + 128, 0, 0);  // global_prefetch_b8
    float d = row[j];
    if (useDist) d = 2.f * d - sqi - sqb[j];
    if (d > vmin) {
#pragma unroll
      for (int t = 0; t < K_NN; ++t) if (t == amin) { vals[t] = d; ids[t] = j; }
      vmin = vals[0]; amin = 0;
#pragma unroll
      for (int t = 1; t < K_NN; ++t) if (vals[t] < vmin) { vmin = vals[t]; amin = t; }
    }
  }
  int* outp = idx + (size_t)tid * K_NN;
#pragma unroll
  for (int t = 0; t < K_NN; ++t) outp[t] = ids[t];
}

// ---------------------------------------------------------------------------
// Fused edge-conv. Column = k*16 + n so each 16-col WMMA tile is one neighbor
// k -> max over K is elementwise max of 20 accumulator fragments.
// The center half of the features is k-invariant, so its contribution
// (W_hi x center) is accumulated ONCE into accC (single WMMA per chunk,
// coalesced staging, no gather) and folded in at the epilogue:
//   y[t] = acc[t] + accC;  max/sum/sumsq adjust accordingly.
// ---------------------------------------------------------------------------
template <int CFEAT, int CIN, int COUT>
__global__ __launch_bounds__(32)
void edge_conv_wmma_kernel(const float* __restrict__ xc, const float* __restrict__ src,
                           const int* __restrict__ idx, const float* __restrict__ W,
                           float* __restrict__ ymax, float* __restrict__ sum,
                           float* __restrict__ sumsq) {
  __shared__ __align__(16) _Float16 Fs[NCOL][CPAD];   // gathered neigh-center (f16)
  __shared__ __align__(16) float    Cf[16][FPAD];     // center chunk (f32)
  __shared__ __align__(16) _Float16 Bs[16][CPAD];     // center chunk (f16, phase 2)
  const int lane = threadIdx.x;
  const int m = lane & 15, hi = lane >> 4;
  const int n0 = blockIdx.x * 16;
  const int ot = blockIdx.y;
  const int b  = blockIdx.z;
  const float* xb = xc  + (size_t)b * CFEAT * N_PTS;
  const float* sb = src + (size_t)b * CFEAT * N_PTS;
  const int*   ib = idx + (size_t)b * N_PTS * K_NN;
  const int orow = ot * 16 + m;

  v8f acc[K_NN], accC;
#pragma unroll
  for (int t = 0; t < K_NN; ++t)
#pragma unroll
    for (int r = 0; r < 8; ++r) acc[t][r] = 0.f;
#pragma unroll
  for (int r = 0; r < 8; ++r) accC[r] = 0.f;

  if constexpr (CIN % 32 == 0) {
    // ---------- Phase 1: neighbor - center chunks (gathered) ----------
#pragma unroll
    for (int c0 = 0; c0 < CFEAT; c0 += 32) {
      stage_tile_f32(xb + (size_t)c0 * N_PTS + n0, lane, Cf);
      __syncthreads();
#pragma unroll
      for (int cl = 0; cl < NCOL / 32; ++cl) {
        int col = cl * 32 + lane;
        int k = col >> 4, nl = col & 15;
        int j = ib[(size_t)(n0 + nl) * K_NN + k];
        const float* sp = sb + (size_t)c0 * N_PTS + j;
        const float* crow = &Cf[nl][0];
#pragma unroll
        for (int r = 0; r < 32; r += 2) {
          float va = sp[(size_t)r * N_PTS]       - crow[r];
          float vb = sp[(size_t)(r + 1) * N_PTS] - crow[r + 1];
          v2h pk; pk[0] = (_Float16)va; pk[1] = (_Float16)vb;
          *(v2h*)&Fs[col][r] = pk;          // packed ds_store_b32
        }
      }
      __syncthreads();
      v16h a = load_w_tile<CIN>(W, orow, c0, hi);
#pragma unroll
      for (int t = 0; t < K_NN; ++t)
        acc[t] = wmma_f16(a, frag_b_from_lds(&Fs[t * 16], m, hi), acc[t]);
      __syncthreads();
    }
    // ---------- Phase 2: center chunks, k-invariant: one WMMA per chunk ----
#pragma unroll
    for (int c0 = 0; c0 < CFEAT; c0 += 32) {
      stage_tile(xb + (size_t)c0 * N_PTS + n0, lane, Bs);
      __syncthreads();
      v16h a = load_w_tile<CIN>(W, orow, CFEAT + c0, hi);
      accC = wmma_f16(a, frag_b_from_lds(Bs, m, hi), accC);
      __syncthreads();
    }
  } else {
    // ---------- Stage 1 (CFEAT=3, CIN=6): single mixed chunk ----------
#pragma unroll
    for (int cl = 0; cl < NCOL / 32; ++cl) {
      int col = cl * 32 + lane;
      int k = col >> 4, nl = col & 15;
      int n = n0 + nl;
      int j = ib[(size_t)n * K_NN + k];
#pragma unroll
      for (int r = 0; r < 32; r += 2) {
        float pv[2];
#pragma unroll
        for (int s = 0; s < 2; ++s) {
          const int c = r + s;                  // compile-time after unroll
          float v;
          if (c >= CIN)        v = 0.f;
          else if (c < CFEAT)  v = sb[(size_t)c * N_PTS + j] - xb[(size_t)c * N_PTS + n];
          else                 v = xb[(size_t)(c - CFEAT) * N_PTS + n];
          pv[s] = v;
        }
        v2h pk; pk[0] = (_Float16)pv[0]; pk[1] = (_Float16)pv[1];
        *(v2h*)&Fs[col][r] = pk;
      }
    }
    __syncthreads();
    // Narrow weight: clamped address + value mask, no exec-divergent loads.
    v16h a;
#pragma unroll
    for (int p = 0; p < 16; ++p) {
      int ka = ((p < 8) ? p : p + 8) + hi * 8;
      int kc = (ka < CIN) ? ka : 0;
      float v = W[(size_t)orow * CIN + kc];
      a[p] = (_Float16)((ka < CIN) ? v : 0.f);
    }
#pragma unroll
    for (int t = 0; t < K_NN; ++t)
      acc[t] = wmma_f16(a, frag_b_from_lds(&Fs[t * 16], m, hi), acc[t]);
  }

  // ---------- Epilogue: fold accC, max over K, BN stats ----------
#pragma unroll
  for (int r = 0; r < 8; ++r) {
    int o = ot * 16 + r + 8 * hi;
    float base = accC[r];
    float mx = -FLT_MAX, ps = 0.f, pq = 0.f;
#pragma unroll
    for (int t = 0; t < K_NN; ++t) {
      float v = acc[t][r] + base;
      mx = fmaxf(mx, v); ps += v; pq += v * v;
    }
    ymax[((size_t)b * COUT + o) * N_PTS + n0 + m] = mx;
    atomicAdd(&sum[o], ps);
    atomicAdd(&sumsq[o], pq);
  }
}

// ---------------------------------------------------------------------------
// BN stats -> affine; apply with LeakyReLU.
// ---------------------------------------------------------------------------
__global__ void finalize_stats_kernel(const float* __restrict__ sum,
                                      const float* __restrict__ sumsq,
                                      const float* __restrict__ g,
                                      const float* __restrict__ beta,
                                      float* __restrict__ scale,
                                      float* __restrict__ shift,
                                      int Cout, float inv_count) {
  int o = blockIdx.x * blockDim.x + threadIdx.x;
  if (o >= Cout) return;
  float mean = sum[o] * inv_count;
  float var  = sumsq[o] * inv_count - mean * mean;
  float sc   = g[o] * rsqrtf(var + 1e-5f);
  scale[o] = sc;
  shift[o] = beta[o] - mean * sc;
}

__global__ void bn_lrelu_kernel(const float* __restrict__ yin,
                                const float* __restrict__ scale,
                                const float* __restrict__ shift,
                                float* __restrict__ out, int Cout) {
  size_t tid = (size_t)blockIdx.x * blockDim.x + threadIdx.x;
  size_t total = (size_t)B_SZ * Cout * N_PTS;
  if (tid >= total) return;
  int o = (int)((tid / N_PTS) % (size_t)Cout);
  float v = yin[tid] * scale[o] + shift[o];
  out[tid] = (v >= 0.f) ? v : 0.2f * v;
}

// ---------------------------------------------------------------------------
// Final 512x512 pointwise conv on concat[x1..x4]; segment choice is
// compile-time (chunk loop fully unrolled, boundaries 32-aligned).
// ---------------------------------------------------------------------------
__global__ __launch_bounds__(32)
void gemm5_wmma_kernel(const float* __restrict__ x1, const float* __restrict__ x2,
                       const float* __restrict__ x3, const float* __restrict__ x4,
                       const float* __restrict__ W5, float* __restrict__ y5,
                       float* __restrict__ sum, float* __restrict__ sumsq) {
  __shared__ __align__(16) _Float16 Cs[16][CPAD];
  const int lane = threadIdx.x;
  const int m = lane & 15, hi = lane >> 4;
  const int j0 = blockIdx.x * 16, ot = blockIdx.y, b = blockIdx.z;
  v8f acc;
#pragma unroll
  for (int r = 0; r < 8; ++r) acc[r] = 0.f;
#pragma unroll
  for (int c0 = 0; c0 < 512; c0 += 32) {
    const float* segp;
    if (c0 < 64)       segp = x1 + ((size_t)b * 64  +  c0)        * N_PTS;
    else if (c0 < 128) segp = x2 + ((size_t)b * 64  + (c0 - 64))  * N_PTS;
    else if (c0 < 256) segp = x3 + ((size_t)b * 128 + (c0 - 128)) * N_PTS;
    else               segp = x4 + ((size_t)b * 256 + (c0 - 256)) * N_PTS;
    stage_tile(segp + j0, lane, Cs);
    __syncthreads();
    v16h a  = load_w_tile<512>(W5, ot * 16 + m, c0, hi);
    v16h bb = frag_b_from_lds(Cs, m, hi);
    acc = wmma_f16(a, bb, acc);
    __syncthreads();
  }
#pragma unroll
  for (int r = 0; r < 8; ++r) {
    int o = ot * 16 + r + 8 * hi;
    float v = acc[r];
    y5[((size_t)b * 512 + o) * N_PTS + j0 + m] = v;
    atomicAdd(&sum[o], v);
    atomicAdd(&sumsq[o], v * v);
  }
}

// ---------------------------------------------------------------------------
extern "C" void kernel_launch(void* const* d_in, const int* in_sizes, int n_in,
                              void* d_out, int out_size, void* d_ws, size_t ws_size,
                              hipStream_t stream) {
  (void)in_sizes; (void)n_in; (void)out_size; (void)ws_size;
  const float* x  = (const float*)d_in[0];
  const float* y  = (const float*)d_in[1];
  const float* w1 = (const float*)d_in[2];
  const float* w2 = (const float*)d_in[3];
  const float* w3 = (const float*)d_in[4];
  const float* w4 = (const float*)d_in[5];
  const float* w5 = (const float*)d_in[6];
  const float* g1 = (const float*)d_in[7];
  const float* b1 = (const float*)d_in[8];
  const float* g2 = (const float*)d_in[9];
  const float* b2 = (const float*)d_in[10];
  const float* g3 = (const float*)d_in[11];
  const float* b3 = (const float*)d_in[12];
  const float* g4 = (const float*)d_in[13];
  const float* b4 = (const float*)d_in[14];
  const float* g5 = (const float*)d_in[15];
  const float* b5 = (const float*)d_in[16];
  float* out = (float*)d_out;

  char* base = (char*)d_ws;
  size_t off = 0;
  auto carve = [&](size_t bytes) -> char* {
    char* p = base + off;
    off = (off + bytes + 255) & ~(size_t)255;
    return p;
  };
  int*   idx   = (int*)  carve((size_t)B_SZ * N_PTS * K_NN * sizeof(int));
  float* sq    = (float*)carve((size_t)B_SZ * N_PTS * sizeof(float));
  float* G     = (float*)carve((size_t)B_SZ * N_PTS * N_PTS * sizeof(float));
  float* x1    = (float*)carve((size_t)B_SZ * 64  * N_PTS * sizeof(float));
  float* x2    = (float*)carve((size_t)B_SZ * 64  * N_PTS * sizeof(float));
  float* x3    = (float*)carve((size_t)B_SZ * 128 * N_PTS * sizeof(float));
  float* x4    = (float*)carve((size_t)B_SZ * 256 * N_PTS * sizeof(float));
  float* ymaxb = (float*)carve((size_t)B_SZ * 256 * N_PTS * sizeof(float));
  float* xn    = (float*)carve((size_t)B_SZ * 128 * N_PTS * sizeof(float));
  float* yn    = (float*)carve((size_t)B_SZ * 128 * N_PTS * sizeof(float));
  float* y5    = (float*)carve((size_t)B_SZ * 512 * N_PTS * sizeof(float));
  float* ssum  = (float*)carve(512 * sizeof(float));
  float* ssq   = (float*)carve(512 * sizeof(float));
  float* scale = (float*)carve(512 * sizeof(float));
  float* shift = (float*)carve(512 * sizeof(float));

  const dim3 b32(32, 1, 1);
  const dim3 gGram(N_PTS / 16, N_PTS / 16, B_SZ);
  const float invNK = 1.f / (float)((size_t)B_SZ * N_PTS * K_NN);
  const float invN  = 1.f / (float)((size_t)B_SZ * N_PTS);

  // ---------------- Stage 1: C=3 -> 64 (LDS-resident kNN) ----------------
  knn3_kernel<<<dim3(N_PTS / 256, B_SZ), 256, 0, stream>>>(x, idx);
  hipMemsetAsync(ssum, 0, 64 * sizeof(float), stream);
  hipMemsetAsync(ssq,  0, 64 * sizeof(float), stream);
  edge_conv_wmma_kernel<3, 6, 64><<<dim3(N_PTS / 16, 4, B_SZ), b32, 0, stream>>>(
      x, x, idx, w1, ymaxb, ssum, ssq);
  finalize_stats_kernel<<<1, 256, 0, stream>>>(ssum, ssq, g1, b1, scale, shift, 64, invNK);
  bn_lrelu_kernel<<<((size_t)B_SZ * 64 * N_PTS + 255) / 256, 256, 0, stream>>>(
      ymaxb, scale, shift, x1, 64);

  // ---------------- Stage 2: 64 -> 64 ----------------
  sqnorm_kernel<<<32, 256, 0, stream>>>(x1, sq, 64);
  gram_wmma_kernel<64><<<gGram, b32, 0, stream>>>(x1, x1, G);
  topk_kernel<<<32, 256, 0, stream>>>(G, sq, idx, 1);
  hipMemsetAsync(ssum, 0, 64 * sizeof(float), stream);
  hipMemsetAsync(ssq,  0, 64 * sizeof(float), stream);
  edge_conv_wmma_kernel<64, 128, 64><<<dim3(N_PTS / 16, 4, B_SZ), b32, 0, stream>>>(
      x1, x1, idx, w2, ymaxb, ssum, ssq);
  finalize_stats_kernel<<<1, 256, 0, stream>>>(ssum, ssq, g2, b2, scale, shift, 64, invNK);
  bn_lrelu_kernel<<<((size_t)B_SZ * 64 * N_PTS + 255) / 256, 256, 0, stream>>>(
      ymaxb, scale, shift, x2, 64);

  // ---------------- Stage 3: 64 -> 128 ----------------
  sqnorm_kernel<<<32, 256, 0, stream>>>(x2, sq, 64);
  gram_wmma_kernel<64><<<gGram, b32, 0, stream>>>(x2, x2, G);
  topk_kernel<<<32, 256, 0, stream>>>(G, sq, idx, 1);
  hipMemsetAsync(ssum, 0, 128 * sizeof(float), stream);
  hipMemsetAsync(ssq,  0, 128 * sizeof(float), stream);
  edge_conv_wmma_kernel<64, 128, 128><<<dim3(N_PTS / 16, 8, B_SZ), b32, 0, stream>>>(
      x2, x2, idx, w3, ymaxb, ssum, ssq);
  finalize_stats_kernel<<<1, 256, 0, stream>>>(ssum, ssq, g3, b3, scale, shift, 128, invNK);
  bn_lrelu_kernel<<<((size_t)B_SZ * 128 * N_PTS + 255) / 256, 256, 0, stream>>>(
      ymaxb, scale, shift, x3, 128);

  // ---------------- Stage 4: cosine kNN(x3, y), 128 -> 256 ----------------
  normalize_kernel<<<32, 256, 0, stream>>>(x3, xn, 128);
  normalize_kernel<<<32, 256, 0, stream>>>(y,  yn, 128);
  gram_wmma_kernel<128><<<gGram, b32, 0, stream>>>(xn, yn, G);
  topk_kernel<<<32, 256, 0, stream>>>(G, sq, idx, 0);   // similarity mode
  hipMemsetAsync(ssum, 0, 256 * sizeof(float), stream);
  hipMemsetAsync(ssq,  0, 256 * sizeof(float), stream);
  edge_conv_wmma_kernel<128, 256, 256><<<dim3(N_PTS / 16, 16, B_SZ), b32, 0, stream>>>(
      x3, y, idx, w4, ymaxb, ssum, ssq);
  finalize_stats_kernel<<<1, 256, 0, stream>>>(ssum, ssq, g4, b4, scale, shift, 256, invNK);
  bn_lrelu_kernel<<<((size_t)B_SZ * 256 * N_PTS + 255) / 256, 256, 0, stream>>>(
      ymaxb, scale, shift, x4, 256);

  // ---------------- Stage 5: concat -> 512x512 conv ----------------
  hipMemsetAsync(ssum, 0, 512 * sizeof(float), stream);
  hipMemsetAsync(ssq,  0, 512 * sizeof(float), stream);
  gemm5_wmma_kernel<<<dim3(N_PTS / 16, 32, B_SZ), b32, 0, stream>>>(
      x1, x2, x3, x4, w5, y5, ssum, ssq);
  finalize_stats_kernel<<<2, 256, 0, stream>>>(ssum, ssq, g5, b5, scale, shift, 512, invN);
  bn_lrelu_kernel<<<((size_t)B_SZ * 512 * N_PTS + 255) / 256, 256, 0, stream>>>(
      y5, scale, shift, out, 512);
}
